// Generator_lstm_8675833938264
// MI455X (gfx1250) — matrix operand order repaired and checked by reading, not compiled, not run on hardware
//
#include <hip/hip_runtime.h>
#include <hip/hip_bf16.h>
#include <stdint.h>

// ---------------------------------------------------------------------------
// Problem constants (from reference)
// ---------------------------------------------------------------------------
#define BB   128          // batch
#define TT   2048         // time
#define CC   16           // input channels
#define CO   128          // conv out channels (= LSTM input)
#define HH   128          // hidden
#define GG   512          // 4*H gate dim
#define NEG_SLOPE 0.01f

// workspace layout (bytes)
//   x0 image   : [btile(8)][t(2048)][gtile(32)][lane(32)] * 16B = 256 MiB
//   weight imgs: 4 x 128 KiB f16 B-fragment images, canonical layout
//                half index = ((row_tile*4 + s)*32 + lane)*16 + e
//                order: [0]=w_ih0 [1]=w_ih1 [2]=w_hh1 [3]=w_hh0
//   conv image : 16 KiB f16 B-fragment image of conv_w, K padded 48->64
#define X0_BYTES   ((size_t)8 * 2048 * 32 * 32 * 16)
#define WIMG_OFF   X0_BYTES
#define WIMG_ONE   (131072u)          // bytes per weight image (512*128 halfs)
#define CWIMG_HALF_OFF (4u * 65536u)  // half offset of conv image in img[]

typedef __attribute__((ext_vector_type(16))) _Float16 v16h;
typedef __attribute__((ext_vector_type(8)))  float    v8f;

union H8U4 { uint4 u; _Float16 h[8]; };
union H16U { uint4 u[2]; v16h v; };

__device__ __forceinline__ float sigmf(float x) {
    return 1.0f / (1.0f + __expf(-x));
}

// A-fragment (16x32 f16, M x K) from a row-major f16 [16][stride] LDS tile.
// Lane l: M = l&15 ; K = 32*s + kb + {0..15},
// kb = (l>=16) ? 16 : 0.   (contiguous order, matches B images)
__device__ __forceinline__ v16h load_a_frag(const _Float16* tile, int stride,
                                            int s, int lane) {
    int m  = lane & 15;
    int kb = (lane >> 4) << 4;
    const _Float16* p = tile + m * stride + s * 32 + kb;
    H16U r;
    r.u[0] = *(const uint4*)(p);
    r.u[1] = *(const uint4*)(p + 8);
    return r.v;
}

// ---------------------------------------------------------------------------
// Kernel P: pre-pack the four weight matrices + conv_w into f16 B-fragment
// images.  B[k][n] = W[rt*16+n][k]; lane l: N=l&15, K=32*s+((l>=16)?16:0)+e.
// conv_w image: N = out-channel, K = c*3+dt (48 taps), zero-padded to 64.
// ---------------------------------------------------------------------------
__global__ void prep_weight_images(const float* __restrict__ w_ih0,
                                   const float* __restrict__ w_ih1,
                                   const float* __restrict__ w_hh1,
                                   const float* __restrict__ w_hh0,
                                   const float* __restrict__ conv_w,
                                   _Float16* __restrict__ img) {
    int idx = blockIdx.x * blockDim.x + threadIdx.x;   // 0 .. 270335
    if (idx < 4 * 65536) {
        int sel   = idx >> 16;        // which weight
        int inner = idx & 65535;
        int e    = inner & 15;
        int lane = (inner >> 4) & 31;
        int s    = (inner >> 9) & 3;
        int rt   = (inner >> 11) & 31;   // row tile 0..31
        int row  = rt * 16 + (lane & 15);
        int col  = s * 32 + ((lane >> 4) << 4) + e;
        const float* W = (sel == 0) ? w_ih0 : (sel == 1) ? w_ih1
                       : (sel == 2) ? w_hh1 : w_hh0;
        img[(size_t)sel * 65536 + inner] = (_Float16)W[(size_t)row * 128 + col];
    } else if (idx < 4 * 65536 + 8192) {
        int inner = idx - 4 * 65536;     // conv image, 8 ct * 2 s * 32 * 16
        int e    = inner & 15;
        int lane = (inner >> 4) & 31;
        int s    = (inner >> 9) & 1;
        int ct   = (inner >> 10) & 7;    // out-channel tile 0..7
        int co   = ct * 16 + (lane & 15);
        int k    = s * 32 + ((lane >> 4) << 4) + e;   // 0..63
        float v  = (k < 48) ? conv_w[(size_t)co * 48 + k] : 0.0f;
        img[CWIMG_HALF_OFF + inner] = (_Float16)v;
    }
}

// ---------------------------------------------------------------------------
// Kernel X0: conv1d(k=3,pad=1) + LeakyReLU + x@w_ih0^T + (b_ih0+b_hh0),
// both contractions on WMMA.  One block (128 thr / 4 waves) per
// (batch-tile of 16, time step t).
// Stage 1: A = x window [16 x 64] (K = c*3+dt, padded), B = conv image.
// Stage 2: A = seq [16 x 128], B = w_ih0 fragment image.
// Output: f16 in WMMA C-tile layout -> 16B per lane per gate tile.
// ---------------------------------------------------------------------------
__global__ void __launch_bounds__(128)
x0_conv_gemm(const float* __restrict__ x,
             const v16h* __restrict__ cwf,        // conv fragment image [8][2][32]
             const float* __restrict__ conv_b,
             const v16h* __restrict__ wih0f,      // fragment image [32][4][32]
             const float* __restrict__ b_ih0,
             const float* __restrict__ b_hh0,
             char* __restrict__ x0img) {
    __shared__ _Float16 axb[16][64];     // conv A tile: [row][c*3+dt], pad->0
    __shared__ _Float16 seq[16][128];    // conv output tile (stage-2 A)

    const int t     = blockIdx.x;
    const int btile = blockIdx.y;
    const int tid   = threadIdx.x;
    const int wave  = tid >> 5;          // 0..3
    const int lane  = tid & 31;

    // build conv A tile (zero padding at t=0 / t=T-1 and K 48..63)
    for (int i = tid; i < 16 * 64; i += 128) {
        int r = i >> 6;
        int k = i & 63;
        float v = 0.0f;
        if (k < 48) {
            int c  = k / 3;
            int dt = k - c * 3;
            int b  = btile * 16 + r;
            int ts = t + dt - 1;
            if (ts >= 0 && ts < TT) v = x[((size_t)b * TT + ts) * CC + c];
        }
        axb[r][k] = (_Float16)v;
    }
    __syncthreads();

    // stage 1: conv as WMMA.  Wave owns out-channel tiles 2*wave, 2*wave+1.
    {
        v16h ac[2];
#pragma unroll
        for (int s = 0; s < 2; ++s) ac[s] = load_a_frag(&axb[0][0], 64, s, lane);
        int mof = (lane >> 4) << 3;
#pragma unroll
        for (int j = 0; j < 2; ++j) {
            int ct   = wave * 2 + j;
            int ccol = ct * 16 + (lane & 15);
            float cb = conv_b[ccol];
            v8f acc;
#pragma unroll
            for (int e = 0; e < 8; ++e) acc[e] = cb;
#pragma unroll
            for (int s = 0; s < 2; ++s)
                acc = __builtin_amdgcn_wmma_f32_16x16x32_f16(
                    false, ac[s], false, cwf[((size_t)ct * 2 + s) * 32 + lane],
                    (short)0, acc, false, false);
#pragma unroll
            for (int r = 0; r < 8; ++r) {
                float y = acc[r];
                y = (y >= 0.0f) ? y : NEG_SLOPE * y;     // LeakyReLU
                seq[r + mof][ccol] = (_Float16)y;
            }
        }
    }
    __syncthreads();

    // stage 2: input projection.  Wave owns gate tiles wave*8 .. wave*8+7.
    v16h a[4];
#pragma unroll
    for (int s = 0; s < 4; ++s) a[s] = load_a_frag(&seq[0][0], 128, s, lane);

    char* base = x0img + (((size_t)btile * TT + t) * 32) * (32 * 16);

#pragma unroll 2
    for (int j = 0; j < 8; ++j) {
        int gt   = wave * 8 + j;
        int bcol = gt * 16 + (lane & 15);
        float bias = b_ih0[bcol] + b_hh0[bcol];
        v8f acc;
#pragma unroll
        for (int e = 0; e < 8; ++e) acc[e] = bias;
#pragma unroll
        for (int s = 0; s < 4; ++s) {
            v16h b = wih0f[((size_t)gt * 4 + s) * 32 + lane];
            acc = __builtin_amdgcn_wmma_f32_16x16x32_f16(
                false, a[s], false, b, (short)0, acc, false, false);
        }
        H8U4 pk;
#pragma unroll
        for (int e = 0; e < 8; ++e) pk.h[e] = (_Float16)acc[e];
        *(uint4*)(base + ((size_t)gt * 32 + lane) * 16) = pk.u;
    }
}

// ---------------------------------------------------------------------------
// Kernel LSTM: persistent. 8 blocks x 256 thr (8 waves). Block = 16 batch
// rows for all 2048 steps. Wave wv owns hidden slice [16wv,16wv+16) ->
// gate tiles q*8+wv. w_hh0 fragments live in VGPRs the whole loop;
// w_ih1/w_hh1 fragment images live in LDS (TDM copy); h0/h1 staged in LDS.
// ---------------------------------------------------------------------------
#define L_WIH1 0u
#define L_WHH1 131072u
#define L_H0   262144u
#define L_H1   266240u
#define L_RED  270336u
#define L_SIZE 270400u

__global__ void __launch_bounds__(256)
lstm_persistent(const char* __restrict__ x0img,
                const _Float16* __restrict__ wimg,   // base of weight images
                const float* __restrict__ b_ih1,
                const float* __restrict__ b_hh1,
                const float* __restrict__ lin_w,
                const float* __restrict__ lin_b,
                float* __restrict__ out) {
    extern __shared__ char smem[];
    _Float16* h0st = (_Float16*)(smem + L_H0);
    _Float16* h1st = (_Float16*)(smem + L_H1);
    float*    red  = (float*)(smem + L_RED);

    const int btile = blockIdx.x;
    const int tid   = threadIdx.x;
    const int wv    = tid >> 5;      // 0..7
    const int lane  = tid & 31;

    // images: [0]=w_ih0 (unused here) [1]=w_ih1 [2]=w_hh1 [3]=w_hh0
    const _Float16* ih1_src = wimg + (size_t)1 * 65536;
    const v16h*     whh0f   = (const v16h*)(wimg + (size_t)3 * 65536);

    // ---- bring w_ih1/w_hh1 fragment images into LDS (TDM if available) ----
#if __has_builtin(__builtin_amdgcn_tensor_load_to_lds)
    if (wv == 0) {
        typedef unsigned int u32x4 __attribute__((ext_vector_type(4)));
        typedef int i32x4 __attribute__((ext_vector_type(4)));
        typedef int i32x8 __attribute__((ext_vector_type(8)));
        unsigned long long ga = (unsigned long long)(const void*)ih1_src;
        unsigned int lds0 = (unsigned int)(unsigned long long)(smem + L_WIH1);
#pragma unroll
        for (int i = 0; i < 2; ++i) {
            unsigned long long g = ga + (unsigned long long)i * WIMG_ONE;
            u32x4 g0;
            g0[0] = 1u;                                 // count=1
            g0[1] = lds0 + (unsigned)i * WIMG_ONE;      // lds_addr
            g0[2] = (unsigned int)g;                    // global_addr lo
            g0[3] = (unsigned int)(g >> 32) | (2u << 30); // addr hi | type=2
            i32x8 g1;
            g1[0] = 0x00030000;      // data_size = 3 (8B elements)
            g1[1] = (int)0x40000000; // tensor_dim0 = 16384 (bits 63:48)
            g1[2] = 0x00010000;      // tensor_dim1 = 1
            g1[3] = 0x40000000;      // tile_dim0  = 16384 (bits 127:112)
            g1[4] = 0x00000001;      // tile_dim1  = 1
            g1[5] = 16384;           // tensor_dim0_stride
            g1[6] = 0; g1[7] = 0;
            i32x4 z4; z4[0] = 0; z4[1] = 0; z4[2] = 0; z4[3] = 0;
            i32x8 z8;
#pragma unroll
            for (int k = 0; k < 8; ++k) z8[k] = 0;
            __builtin_amdgcn_tensor_load_to_lds(g0, g1, z4, z4, z8, 0);
        }
#if __has_builtin(__builtin_amdgcn_s_wait_tensorcnt)
        __builtin_amdgcn_s_wait_tensorcnt(0);
#endif
    }
#else
    for (unsigned i = tid; i < (2u * WIMG_ONE) / 16u; i += blockDim.x)
        ((uint4*)(smem + L_WIH1))[i] = ((const uint4*)ih1_src)[i];
#endif

    // ---- init h0/h1 staging + reduction buffer ----
    for (int i = tid; i < 2048; i += 256) {
        h0st[i] = (_Float16)0.0f;
        h1st[i] = (_Float16)0.0f;
    }
    if (tid < 16) red[tid] = 0.0f;

    // ---- w_hh0 B-fragments -> registers (held across whole T loop) ----
    v16h bhh0[4][4];
#pragma unroll
    for (int q = 0; q < 4; ++q)
#pragma unroll
        for (int s = 0; s < 4; ++s)
            bhh0[q][s] = whh0f[(((size_t)(q * 8 + wv)) * 4 + s) * 32 + lane];

    float b1q[4];
#pragma unroll
    for (int q = 0; q < 4; ++q) {
        int col = q * 128 + wv * 16 + (lane & 15);
        b1q[q] = b_ih1[col] + b_hh1[col];
    }

    const v16h* wih1f = (const v16h*)(smem + L_WIH1);
    const v16h* whh1f = (const v16h*)(smem + L_WHH1);
    const char* x0b   = x0img + (size_t)btile * TT * 32 * 32 * 16;

    v8f c0, c1, hsum;
#pragma unroll
    for (int e = 0; e < 8; ++e) { c0[e] = 0.0f; c1[e] = 0.0f; hsum[e] = 0.0f; }

    __syncthreads();

    for (int t = 0; t < TT; ++t) {
        // (a) snapshot h0_prev / h1_prev as A fragments
        v16h a0[4], a1[4];
#pragma unroll
        for (int s = 0; s < 4; ++s) {
            a0[s] = load_a_frag(h0st, 128, s, lane);
            a1[s] = load_a_frag(h1st, 128, s, lane);
        }
        __syncthreads();

        // (b) layer 0:  gates = x0[t] + h0_prev @ w_hh0^T
        v8f g0v[4];
#pragma unroll
        for (int q = 0; q < 4; ++q) {
            int gt = q * 8 + wv;
            H8U4 xt;
            xt.u = *(const uint4*)(x0b + (((size_t)t * 32 + gt) * 32 + lane) * 16);
            v8f acc;
#pragma unroll
            for (int e = 0; e < 8; ++e) acc[e] = (float)xt.h[e];
#pragma unroll
            for (int s = 0; s < 4; ++s)
                acc = __builtin_amdgcn_wmma_f32_16x16x32_f16(
                    false, a0[s], false, bhh0[q][s], (short)0, acc, false, false);
            g0v[q] = acc;
        }
        // LSTM cell 0 (elementwise in C layout) -> write h0_new to staging
        {
            int col = wv * 16 + (lane & 15);
            int mof = (lane >> 4) << 3;
#pragma unroll
            for (int r = 0; r < 8; ++r) {
                float ig = sigmf(g0v[0][r]);
                float fg = sigmf(g0v[1][r]);
                float gg = tanhf(g0v[2][r]);
                float og = sigmf(g0v[3][r]);
                c0[r] = fg * c0[r] + ig * gg;
                float hv = og * tanhf(c0[r]);
                h0st[(r + mof) * 128 + col] = (_Float16)hv;
            }
        }
        __syncthreads();

        // (c) layer 1:  gates = b1 + h0_new @ w_ih1^T + h1_prev @ w_hh1^T
        v16h a0n[4];
#pragma unroll
        for (int s = 0; s < 4; ++s) a0n[s] = load_a_frag(h0st, 128, s, lane);

        v8f g1v[4];
#pragma unroll
        for (int q = 0; q < 4; ++q) {
            int gt = q * 8 + wv;
            v8f acc;
#pragma unroll
            for (int e = 0; e < 8; ++e) acc[e] = b1q[q];
#pragma unroll
            for (int s = 0; s < 4; ++s)
                acc = __builtin_amdgcn_wmma_f32_16x16x32_f16(
                    false, a0n[s], false,
                    wih1f[((size_t)gt * 4 + s) * 32 + lane],
                    (short)0, acc, false, false);
#pragma unroll
            for (int s = 0; s < 4; ++s)
                acc = __builtin_amdgcn_wmma_f32_16x16x32_f16(
                    false, a1[s], false,
                    whh1f[((size_t)gt * 4 + s) * 32 + lane],
                    (short)0, acc, false, false);
            g1v[q] = acc;
        }
        {
            int col = wv * 16 + (lane & 15);
            int mof = (lane >> 4) << 3;
#pragma unroll
            for (int r = 0; r < 8; ++r) {
                float ig = sigmf(g1v[0][r]);
                float fg = sigmf(g1v[1][r]);
                float gg = tanhf(g1v[2][r]);
                float og = sigmf(g1v[3][r]);
                c1[r] = fg * c1[r] + ig * gg;
                float hv = og * tanhf(c1[r]);
                hsum[r] += hv;
                h1st[(r + mof) * 128 + col] = (_Float16)hv;
            }
        }
        __syncthreads();
    }

    // ---- mean pool over T + final linear (LDS reduction) ----
    {
        float lw = lin_w[wv * 16 + (lane & 15)] * (1.0f / (float)TT);
        int mof = (lane >> 4) << 3;
#pragma unroll
        for (int r = 0; r < 8; ++r)
            atomicAdd(&red[r + mof], hsum[r] * lw);
    }
    __syncthreads();
    if (tid < 16) out[btile * 16 + tid] = red[tid] + lin_b[0];
}

// ---------------------------------------------------------------------------
// Host-side launch
// ---------------------------------------------------------------------------
extern "C" void kernel_launch(void* const* d_in, const int* in_sizes, int n_in,
                              void* d_out, int out_size, void* d_ws, size_t ws_size,
                              hipStream_t stream) {
    const float* x      = (const float*)d_in[0];
    const float* conv_w = (const float*)d_in[1];
    const float* conv_b = (const float*)d_in[2];
    const float* w_ih0  = (const float*)d_in[3];
    const float* w_hh0  = (const float*)d_in[4];
    const float* b_ih0  = (const float*)d_in[5];
    const float* b_hh0  = (const float*)d_in[6];
    const float* w_ih1  = (const float*)d_in[7];
    const float* w_hh1  = (const float*)d_in[8];
    const float* b_ih1  = (const float*)d_in[9];
    const float* b_hh1  = (const float*)d_in[10];
    const float* lin_w  = (const float*)d_in[11];
    const float* lin_b  = (const float*)d_in[12];

    char*     ws   = (char*)d_ws;
    char*     x0im = ws;
    _Float16* wimg = (_Float16*)(ws + WIMG_OFF);
    float*    out  = (float*)d_out;

    (void)in_sizes; (void)n_in; (void)out_size; (void)ws_size;

    // 1) pack weight matrices + conv_w into f16 B-fragment images
    prep_weight_images<<<1057, 256, 0, stream>>>(w_ih0, w_ih1, w_hh1, w_hh0,
                                                 conv_w, wimg);

    // 2) conv + LeakyReLU + input projection -> x0 tile image (f16)
    dim3 g1(TT, BB / 16);
    x0_conv_gemm<<<g1, 128, 0, stream>>>(x, (const v16h*)(wimg + CWIMG_HALF_OFF),
                                         conv_b, (const v16h*)wimg,
                                         b_ih0, b_hh0, x0im);

    // 3) persistent 2-layer LSTM + pool + linear
    lstm_persistent<<<BB / 16, 256, L_SIZE, stream>>>(
        x0im, wimg, b_ih1, b_hh1, lin_w, lin_b, out);
}
